// GRUD_67817533604153
// MI455X (gfx1250) — matrix operand order repaired and checked by reading, not compiled, not run on hardware
//
#include <hip/hip_runtime.h>
#include <hip/hip_bf16.h>

typedef __attribute__((ext_vector_type(16))) _Float16 v16h;
typedef __attribute__((ext_vector_type(8)))  _Float16 v8h;
typedef __attribute__((ext_vector_type(4)))  _Float16 v4h;
typedef __attribute__((ext_vector_type(8)))  float    v8f;
typedef __attribute__((ext_vector_type(4)))  float    v4f;

#define B_    256
#define T_    512
#define D_    64
#define H_    128
#define K_    256          // 2*D + H
#define MTILE 16
#define NT    511          // T-1 output steps

// ---- LDS layout (bytes) ----
#define OFF_WZ    0        // 128x256 f16 = 64KB (resident, streamed in phase A)
#define OFF_WR    65536    // 64KB (resident, streamed in phase A)
#define OFF_WH    131072   // 64KB (resident, streamed in phase B)
#define OFF_COMB  196608   // 16 x 256 f16 = 8192 B   [xt | h_pre | mask]
#define OFF_COMB2 204800   // 16 x 256 f16 = 8192 B   [xt | r*h  | mask]
#define OFF_H     212992   // 16 x 128 f32 = 8192 B   hidden state
#define OFF_CONST 221184   // 832 floats   = 3328 B
#define SMEM_BYTES 224512

__device__ __forceinline__ v8f wmma_f16(v16h a, v16h b, v8f c) {
  return __builtin_amdgcn_wmma_f32_16x16x32_f16(false, a, false, b, (short)0, c,
                                                false, false);
}

__device__ __forceinline__ float sigmoidf_(float x) {
  return 1.0f / (1.0f + __expf(-x));
}

// A operand (row-major 16xK f16 in LDS): lane m=lane&15, hi=lane>>4;
// halves 0..7 = K ks*32+hi*8.. ; halves 8..15 = K ks*32+16+hi*8..
__device__ __forceinline__ v16h load_a(const _Float16* arow, int ks) {
  union { v16h v; v8h h[2]; } u;
  u.h[0] = *(const v8h*)(arow + ks * 32);
  u.h[1] = *(const v8h*)(arow + ks * 32 + 16);
  return u.v;
}

// B operand: W stored [n][k] f16 -> halves 0..7 = K ks*32+hi*8.. ; halves 8..15 = K ks*32+16+hi*8..
__device__ __forceinline__ v16h load_b(const _Float16* w, int col, int hi, int ks) {
  union { v16h v; v8h h[2]; } u;
  u.h[0] = *(const v8h*)(w + col * K_ + ks * 32 + hi * 8);
  u.h[1] = *(const v8h*)(w + col * K_ + ks * 32 + 16 + hi * 8);
  return u.v;
}

__global__ void grud_persistent_kernel(
    const float* __restrict__ x,     const float* __restrict__ mask,
    const float* __restrict__ delta, const float* __restrict__ X_mean,
    const float* __restrict__ Wz,    const float* __restrict__ bz,
    const float* __restrict__ Wr,    const float* __restrict__ br,
    const float* __restrict__ Wh,    const float* __restrict__ bh,
    const float* __restrict__ w_gx,  const float* __restrict__ b_gx,
    const float* __restrict__ w_gh,  const float* __restrict__ b_gh,
    float* __restrict__ out) {
  extern __shared__ char smem[];
  _Float16* wz_l    = (_Float16*)(smem + OFF_WZ);
  _Float16* wr_l    = (_Float16*)(smem + OFF_WR);
  _Float16* wh_l    = (_Float16*)(smem + OFF_WH);
  _Float16* comb_l  = (_Float16*)(smem + OFF_COMB);
  _Float16* comb2_l = (_Float16*)(smem + OFF_COMB2);
  float*    h_l     = (float*)(smem + OFF_H);
  float*    cst     = (float*)(smem + OFF_CONST);
  float* bz_l  = cst;        // 128
  float* br_l  = cst + 128;  // 128
  float* bh_l  = cst + 256;  // 128
  float* xm_l  = cst + 384;  // 64
  float* wgx_l = cst + 448;  // 64
  float* bgx_l = cst + 512;  // 64
  float* wgh_l = cst + 576;  // 128
  float* bgh_l = cst + 704;  // 128

  const int tid = threadIdx.x;
  const int b0  = blockIdx.x * MTILE;

  // ---- Prologue: stage weights (f32 -> f16, [k][n] -> [n][k]) + consts ----
  for (int i = tid; i < K_ * H_; i += 256) {
    int k = i >> 7, n = i & 127;
    int dst = n * K_ + k;
    wz_l[dst] = (_Float16)Wz[i];
    wr_l[dst] = (_Float16)Wr[i];
    wh_l[dst] = (_Float16)Wh[i];
  }
  if (tid < 128) {
    bz_l[tid] = bz[tid];  br_l[tid] = br[tid];  bh_l[tid] = bh[tid];
    wgh_l[tid] = w_gh[tid];  bgh_l[tid] = b_gh[tid];
  }
  if (tid < 64) {
    xm_l[tid] = X_mean[tid];  wgx_l[tid] = w_gx[tid];  bgx_l[tid] = b_gx[tid];
  }
  for (int i = tid; i < MTILE * H_; i += 256) h_l[i] = 0.0f;
  __syncthreads();

  const int lane = tid & 31;
  const int wv   = tid >> 5;          // wave id 0..7 -> owns N tile wv
  const int nl   = lane & 15;
  const int hi   = lane >> 4;
  const int col  = wv * 16 + nl;      // output column 0..127
  const _Float16* arow  = comb_l  + nl * K_ + hi * 8;
  const _Float16* arow2 = comb2_l + nl * K_ + hi * 8;

  const float zb = bz_l[col];
  const float rb = br_l[col];
  const float hb = bh_l[col];

  // ---- Serial scan over time ----
  for (int s = 0; s < NT; ++s) {
    const int t = s + 1;

    // ---- Elementwise: decay, impute, h <- gh*h, build comb/comb2 f16 ----
    {
      const int m = tid >> 4;              // 0..15 batch row
      const int b = b0 + m;
      const float d = delta[b * T_ + t];
      const int j0 = (tid & 15) * 4;
      {
        const size_t xi = ((size_t)b * T_ + t) * D_ + j0;
        const v4f xn4 = *(const v4f*)&x[xi];
        const v4f xl4 = *(const v4f*)&x[xi - D_];
        const v4f mk4 = *(const v4f*)&mask[xi];
        v4h xt4, mh4;
#pragma unroll
        for (int jj = 0; jj < 4; ++jj) {
          const int j = j0 + jj;
          const float gx = __expf(-fmaxf(0.0f, d * wgx_l[j] + bgx_l[j]));
          const float xt = mk4[jj] * xn4[jj] +
                           (1.0f - mk4[jj]) * (gx * xl4[jj] + (1.0f - gx) * xm_l[j]);
          xt4[jj] = (_Float16)xt;
          mh4[jj] = (_Float16)mk4[jj];
        }
        *(v4h*)(comb_l  + m * K_ + j0)       = xt4;   // [0:64)    = xt
        *(v4h*)(comb2_l + m * K_ + j0)       = xt4;
        *(v4h*)(comb_l  + m * K_ + 192 + j0) = mh4;   // [192:256) = mask
        *(v4h*)(comb2_l + m * K_ + 192 + j0) = mh4;
        if (t + 1 < T_) {
          __builtin_prefetch(&x[xi + D_], 0, 0);
          __builtin_prefetch(&mask[xi + D_], 0, 0);
        }
      }
      {
        const int n0 = (tid & 15) * 8;
        v8f h8 = *(const v8f*)&h_l[m * H_ + n0];
        v8h hh8;
#pragma unroll
        for (int jj = 0; jj < 8; ++jj) {
          const int n = n0 + jj;
          const float gh = __expf(-fmaxf(0.0f, d * wgh_l[n] + bgh_l[n]));
          const float hp = gh * h8[jj];
          h8[jj]  = hp;
          hh8[jj] = (_Float16)hp;
        }
        *(v8f*)&h_l[m * H_ + n0]       = h8;          // h <- gh*h
        *(v8h*)(comb_l + m * K_ + D_ + n0) = hh8;     // [64:192) = h_pre
      }
    }
    __syncthreads();   // (1) comb complete

    // ---- Phase A: z and r GEMMs, B operands streamed from LDS ----
    v8f cz = {}, cr = {};
#pragma unroll
    for (int ks = 0; ks < 8; ++ks) {
      v16h a = load_a(arow, ks);
      cz = wmma_f16(a, load_b(wz_l, col, hi, ks), cz);
      cr = wmma_f16(a, load_b(wr_l, col, hi, ks), cr);
    }

    // Activation straight after own WMMAs (writes go to comb2, no conflict).
    float zreg[8], hpre[8];
#pragma unroll
    for (int g = 0; g < 8; ++g) {
      const int mm = g + 8 * hi;                 // C layout: M = vgpr + 8*hi
      hpre[g] = h_l[mm * H_ + col];
      zreg[g] = sigmoidf_(cz[g] + zb);
      const float rv = sigmoidf_(cr[g] + rb);
      comb2_l[mm * K_ + D_ + col] = (_Float16)(rv * hpre[g]);   // r*h
    }
    __syncthreads();   // (2) comb2 hidden slice complete

    // ---- Phase B: h_tilde GEMM + gated state update + output store ----
    v8f ch = {};
#pragma unroll
    for (int ks = 0; ks < 8; ++ks) {
      v16h a = load_a(arow2, ks);
      ch = wmma_f16(a, load_b(wh_l, col, hi, ks), ch);
    }
#pragma unroll
    for (int g = 0; g < 8; ++g) {
      const int mm = g + 8 * hi;
      const float ht = tanhf(ch[g] + hb);
      const float hn = (1.0f - zreg[g]) * hpre[g] + zreg[g] * ht;
      h_l[mm * H_ + col] = hn;
      out[((size_t)(b0 + mm) * NT + s) * H_ + col] = hn;
    }
    __syncthreads();   // (3) h_l updated before next step's elementwise
  }
}

extern "C" void kernel_launch(void* const* d_in, const int* in_sizes, int n_in,
                              void* d_out, int out_size, void* d_ws, size_t ws_size,
                              hipStream_t stream) {
  (void)in_sizes; (void)n_in; (void)out_size; (void)d_ws; (void)ws_size;
  const float* x      = (const float*)d_in[0];
  const float* mask   = (const float*)d_in[1];
  const float* delta  = (const float*)d_in[2];
  const float* X_mean = (const float*)d_in[3];
  const float* Wz     = (const float*)d_in[4];
  const float* bz     = (const float*)d_in[5];
  const float* Wr     = (const float*)d_in[6];
  const float* br     = (const float*)d_in[7];
  const float* Wh     = (const float*)d_in[8];
  const float* bh     = (const float*)d_in[9];
  const float* w_gx   = (const float*)d_in[10];
  const float* b_gx   = (const float*)d_in[11];
  const float* w_gh   = (const float*)d_in[12];
  const float* b_gh   = (const float*)d_in[13];
  float* out = (float*)d_out;

  hipFuncSetAttribute((const void*)grud_persistent_kernel,
                      hipFuncAttributeMaxDynamicSharedMemorySize, SMEM_BYTES);
  grud_persistent_kernel<<<dim3(B_ / MTILE), dim3(256), SMEM_BYTES, stream>>>(
      x, mask, delta, X_mean, Wz, bz, Wr, br, Wh, bh, w_gx, b_gx, w_gh, b_gh, out);
}
